// MultiHeadAttention_72670846648350
// MI455X (gfx1250) — matrix operand order repaired and checked by reading, not compiled, not run on hardware
//
#include <hip/hip_runtime.h>
#include <hip/hip_bf16.h>

// ---------------------------------------------------------------------------
// MHA forward for MI455X (gfx1250): bf16 WMMA 16x16x32, f32 accumulate.
// QKV GEMM (64x64/wave) -> flash attention (32 q-rows/wave, TDM K-tile
// loads into LDS) -> output GEMM.
// ---------------------------------------------------------------------------

#define Bc  4
#define Sc  2048
#define Dc  1024
#define Hc  16
#define DKc 64
#define D3c 3072

typedef __bf16 bf16_t;
typedef __attribute__((ext_vector_type(16))) bf16_t v16bf;
typedef __attribute__((ext_vector_type(2)))  bf16_t v2bf;
typedef __attribute__((ext_vector_type(8)))  float  v8f;
typedef __attribute__((ext_vector_type(4)))  unsigned int v4u;
typedef __attribute__((ext_vector_type(8)))  int    v8i;
typedef __attribute__((ext_vector_type(4)))  int    v4i;

#if defined(__has_builtin)
#if __has_builtin(__builtin_amdgcn_tensor_load_to_lds)
#define HAVE_TDM 1
#endif
#endif

union Frag {
    unsigned int u[8];
    v16bf        v;
};

static __device__ __forceinline__ unsigned short f2bf(float f) {
    unsigned int u = __float_as_uint(f);
    u = (u + 0x7FFFu + ((u >> 16) & 1u)) >> 16;   // round-to-nearest-even
    return (unsigned short)u;
}

static __device__ __forceinline__ unsigned int packbf2(float lo, float hi) {
#if __has_builtin(__builtin_amdgcn_cvt_pk_bf16_f32)
    v2bf r = __builtin_amdgcn_cvt_pk_bf16_f32(lo, hi);
    union { v2bf v; unsigned int u; } cv;
    cv.v = r;
    return cv.u;
#else
    return (unsigned int)f2bf(lo) | ((unsigned int)f2bf(hi) << 16);
#endif
}

static __device__ __forceinline__ v8f wmma_bf16(const Frag& a, const Frag& b, v8f c) {
    return __builtin_amdgcn_wmma_f32_16x16x32_bf16(
        false, a.v, false, b.v, (short)0, c, false, false);
}

static __device__ __forceinline__ void wait_tensorcnt0() {
#if __has_builtin(__builtin_amdgcn_s_wait_tensorcnt)
    __builtin_amdgcn_s_wait_tensorcnt(0);
#else
    asm volatile("s_wait_tensorcnt 0x0" ::: "memory");
#endif
    asm volatile("" ::: "memory");   // keep LDS reads below the wait
}

// ---------------------------------------------------------------------------
// Kernel 1: QKV projection.  Y[8192,3072] = X[8192,1024] @ Wqkv^T + b.
// 64x64 output tile per wave = 4x4 WMMA tiles, 16 WMMAs per 32-K step.
// Scatters bf16:  q,k -> [B,H,S,DK]   v -> transposed [B,H,DK,S]
// ---------------------------------------------------------------------------
__global__ __launch_bounds__(256) void mha_qkv_wmma(
    const float* __restrict__ X, const float* __restrict__ W,
    const float* __restrict__ bias,
    unsigned short* __restrict__ qo, unsigned short* __restrict__ ko,
    unsigned short* __restrict__ vto)
{
    const int lane = threadIdx.x & 31;
    const int wid  = (blockIdx.x * blockDim.x + threadIdx.x) >> 5;
    const int NT   = D3c / 64;                 // 48 column tiles
    const int row0 = (wid / NT) * 64;
    const int col0 = (wid % NT) * 64;
    const int mloc = lane & 15;
    const int hsel = (lane >> 4) & 1;          // half-wave select

    v8f acc[16];
#pragma unroll
    for (int t = 0; t < 16; ++t) acc[t] = (v8f){0,0,0,0,0,0,0,0};

    for (int kb = 0; kb < Dc; kb += 32) {
        Frag a[4], b[4];
#pragma unroll
        for (int g = 0; g < 4; ++g) {
            const float* Xr = &X[(size_t)(row0 + g * 16 + mloc) * Dc];
            const float* Wr = &W[(size_t)(col0 + g * 16 + mloc) * Dc];
            __builtin_prefetch(Xr + kb + 32, 0, 0);   // global_prefetch_b8
            __builtin_prefetch(Wr + kb + 32, 0, 0);
#pragma unroll
            for (int r = 0; r < 8; ++r) {
                // A-frag (16-bit 16x32): K = (r/4)*16 + hsel*8 + 2*(r&3)
                const int kA = kb + ((r >> 2) << 4) + (hsel << 3) + ((r & 3) << 1);
                float2 x = *(const float2*)(Xr + kA);
                a[g].u[r] = packbf2(x.x, x.y);
                // B-frag (16-bit 32x16): K = (r/4)*16 + hsel*8 + 2*(r&3), N = mloc
                const int kB = kb + ((r >> 2) << 4) + (hsel << 3) + ((r & 3) << 1);
                float2 w = *(const float2*)(Wr + kB);
                b[g].u[r] = packbf2(w.x, w.y);
            }
        }
#pragma unroll
        for (int mi = 0; mi < 4; ++mi)
#pragma unroll
            for (int ni = 0; ni < 4; ++ni)
                acc[mi * 4 + ni] = wmma_bf16(a[mi], b[ni], acc[mi * 4 + ni]);
    }

#pragma unroll
    for (int t = 0; t < 16; ++t) {
        const int rowt = row0 + ((t >> 2) << 4);
        const int colt = col0 + ((t & 3) << 4);
#pragma unroll
        for (int r = 0; r < 8; ++r) {
            const int m = rowt + r + (hsel << 3);      // global row (b*S+s)
            const int n = colt + mloc;                 // qkv channel
            const float val = acc[t][r] + bias[n];
            const int bI = m >> 11, s = m & (Sc - 1);
            const unsigned short bv = f2bf(val);
            if (n < Dc) {
                const int h = n >> 6, dk = n & 63;
                qo[((size_t)(bI * Hc + h) * Sc + s) * DKc + dk] = bv;
            } else if (n < 2 * Dc) {
                const int c = n - Dc, h = c >> 6, dk = c & 63;
                ko[((size_t)(bI * Hc + h) * Sc + s) * DKc + dk] = bv;
            } else {
                const int c = n - 2 * Dc, h = c >> 6, dk = c & 63;
                vto[((size_t)(bI * Hc + h) * DKc + dk) * Sc + s] = bv;
            }
        }
    }
}

// ---------------------------------------------------------------------------
// Kernel 2: causal flash attention.  One wave per (b,h, 32 query rows).
// Per 32-key step: TDM DMA of the 32x64 bf16 K tile into LDS (TENSORcnt),
// 8 WMMAs (Q@K^T), online softmax (shfl reductions), P relayout D->A via
// wave-private LDS, 8 WMMAs (P@V).
// ---------------------------------------------------------------------------
__global__ __launch_bounds__(256) void mha_attn_wmma(
    const unsigned short* __restrict__ q, const unsigned short* __restrict__ k,
    const unsigned short* __restrict__ vt, unsigned short* __restrict__ obuf)
{
#if HAVE_TDM
    __shared__ unsigned short kbuf[8][32][64];   // per-wave K tile (TDM dest)
#endif
    __shared__ unsigned short pbuf[8][32][32];   // per-wave 32x32 bf16 P tile

    const int lane = threadIdx.x & 31;
    const int w    = threadIdx.x >> 5;
    const int wid  = blockIdx.x * 8 + w;
    const int bh   = wid >> 6;                   // b*H + h  (S/32 = 64 tiles)
    const int qbase = (wid & 63) << 5;           // 32 query rows per wave
    const int mloc = lane & 15;
    const int hsel = (lane >> 4) & 1;

    const unsigned short* qp = q  + (size_t)bh * Sc * DKc;
    const unsigned short* kp = k  + (size_t)bh * Sc * DKc;
    const unsigned short* vp = vt + (size_t)bh * DKc * Sc;

    // Q A-fragments: [row group g][dk half d]
    Frag qa[2][2];
#pragma unroll
    for (int g = 0; g < 2; ++g)
#pragma unroll
        for (int r = 0; r < 8; ++r) {
            const int dk = ((r >> 2) << 4) + (hsel << 3) + ((r & 3) << 1);
            const size_t base = (size_t)(qbase + g * 16 + mloc) * DKc;
            qa[g][0].u[r] = *(const unsigned int*)(qp + base + dk);
            qa[g][1].u[r] = *(const unsigned int*)(qp + base + dk + 32);
        }

    float mi[2][8], li[2][8];
    v8f o[2][4];
#pragma unroll
    for (int g = 0; g < 2; ++g) {
#pragma unroll
        for (int r = 0; r < 8; ++r) { mi[g][r] = -3.0e38f; li[g][r] = 0.0f; }
#pragma unroll
        for (int nt = 0; nt < 4; ++nt) o[g][nt] = (v8f){0,0,0,0,0,0,0,0};
    }

#if HAVE_TDM
    // Wave-uniform pieces of the K-tile tensor descriptor (D#).
    const unsigned lds_off = (unsigned)(uintptr_t)(&kbuf[w][0][0]);
    v8i d1;
    d1[0] = 0x00010000;          // workgroup_mask=0, data_size=1 (2 bytes)
    d1[1] = (DKc & 0xFFFF) << 16;              // tensor_dim0 = 64 (lo 16)
    d1[2] = (Sc & 0xFFFF) << 16;               // dim0 hi | tensor_dim1 = 2048
    d1[3] = (DKc & 0xFFFF) << 16;              // dim1 hi | tile_dim0 = 64
    d1[4] = 32;                                // tile_dim1 = 32, tile_dim2 = 0
    d1[5] = DKc;                               // tensor_dim0_stride = 64
    d1[6] = 0;
    d1[7] = 0;
    const v4i z4 = {0, 0, 0, 0};
#if __has_include(<hip/amd_detail/amd_gfx1250_TDM.h>)
    const v8i z8 = {0, 0, 0, 0, 0, 0, 0, 0};
#endif
#endif

    const int nsteps = (qbase >> 5) + 1;         // keys 0 .. qbase+31
    for (int j = 0; j < nsteps; ++j) {
        const int keyb = j << 5;

#if HAVE_TDM
        // DMA the 32x64 bf16 K tile into this wave's LDS buffer.
        asm volatile("s_wait_dscnt 0x0" ::: "memory");   // LDS reuse (WAR)
        {
            const unsigned long long ga =
                (unsigned long long)(uintptr_t)kp +
                (unsigned long long)keyb * (DKc * 2);
            v4u d0;
            d0.x = 1u;                                   // count = 1
            d0.y = lds_off;                              // lds_addr
            d0.z = (unsigned)ga;                         // global_addr lo
            d0.w = (unsigned)(ga >> 32) | (2u << 30);    // addr hi | type=2
#if __has_include(<hip/amd_detail/amd_gfx1250_TDM.h>)
            __builtin_amdgcn_tensor_load_to_lds(d0, d1, z4, z4, z8, 0);
#else
            __builtin_amdgcn_tensor_load_to_lds(d0, d1, z4, z4, 0);
#endif
        }
        wait_tensorcnt0();
#else
        __builtin_prefetch(kp + (size_t)(keyb + 32 + mloc) * DKc, 0, 0);
#endif

        // K^T B-fragments: B[k=dk][n=key] = K[key][dk]; dk pairs contiguous.
        Frag kbf[2][2];                          // [key tile][dk half]
#pragma unroll
        for (int t = 0; t < 2; ++t) {
#if HAVE_TDM
            const unsigned short* kr = &kbuf[w][t * 16 + mloc][0];
#else
            const unsigned short* kr = kp + (size_t)(keyb + t * 16 + mloc) * DKc;
#endif
#pragma unroll
            for (int r = 0; r < 8; ++r) {
                const int dk0 = ((r >> 2) << 4) + (hsel << 3) + ((r & 3) << 1);
                kbf[t][0].u[r] = *(const unsigned int*)(kr + dk0);
                kbf[t][1].u[r] = *(const unsigned int*)(kr + dk0 + 32);
            }
        }
        v8f sc[2][2];                            // [row group][key tile]
#pragma unroll
        for (int g = 0; g < 2; ++g)
#pragma unroll
            for (int t = 0; t < 2; ++t) {
                v8f s = (v8f){0,0,0,0,0,0,0,0};
                s = wmma_bf16(qa[g][0], kbf[t][0], s);
                s = wmma_bf16(qa[g][1], kbf[t][1], s);
                sc[g][t] = s;
            }

        // Online softmax (rows live per half-wave; reduce over 16 lanes).
#pragma unroll
        for (int g = 0; g < 2; ++g)
#pragma unroll
            for (int r = 0; r < 8; ++r) {
                const int rowg = qbase + g * 16 + r + (hsel << 3);
                const int key0 = keyb + mloc, key1 = key0 + 16;
                float e0 = (key0 <= rowg) ? sc[g][0][r] * 0.125f : -1.0e9f;
                float e1 = (key1 <= rowg) ? sc[g][1][r] * 0.125f : -1.0e9f;
                float mx = fmaxf(e0, e1);
#pragma unroll
                for (int off = 8; off > 0; off >>= 1)
                    mx = fmaxf(mx, __shfl_xor(mx, off, 16));
                const float mnew = fmaxf(mi[g][r], mx);
                const float al   = __expf(mi[g][r] - mnew);
                const float p0   = __expf(e0 - mnew);
                const float p1   = __expf(e1 - mnew);
                float rs = p0 + p1;
#pragma unroll
                for (int off = 8; off > 0; off >>= 1)
                    rs += __shfl_xor(rs, off, 16);
                li[g][r] = li[g][r] * al + rs;
                mi[g][r] = mnew;
                o[g][0][r] *= al; o[g][1][r] *= al;
                o[g][2][r] *= al; o[g][3][r] *= al;
                const int rml = g * 16 + r + (hsel << 3);
                pbuf[w][rml][mloc]      = f2bf(p0);
                pbuf[w][rml][mloc + 16] = f2bf(p1);
            }
        asm volatile("s_wait_dscnt 0x0" ::: "memory");   // LDS RAW within wave

        // P A-fragments from LDS (row-major 32x32 bf16).
        Frag pa[2];
#pragma unroll
        for (int g = 0; g < 2; ++g)
#pragma unroll
            for (int r = 0; r < 8; ++r) {
                const int kc = ((r >> 2) << 4) + (hsel << 3) + ((r & 3) << 1);
                pa[g].u[r] = *(const unsigned int*)&pbuf[w][g * 16 + mloc][kc];
            }

        // V B-fragments from transposed V: key pairs contiguous.
        __builtin_prefetch(vp + (size_t)mloc * Sc + keyb + 32, 0, 0);
#pragma unroll
        for (int nt = 0; nt < 4; ++nt) {
            Frag vb;
#pragma unroll
            for (int r = 0; r < 8; ++r) {
                const int kk = ((r >> 2) << 4) + (hsel << 3) + ((r & 3) << 1);
                vb.u[r] = *(const unsigned int*)
                    (vp + (size_t)(nt * 16 + mloc) * Sc + keyb + kk);
            }
            o[0][nt] = wmma_bf16(pa[0], vb, o[0][nt]);
            o[1][nt] = wmma_bf16(pa[1], vb, o[1][nt]);
        }
    }

    // Normalize and scatter bf16 O into [B,S,D] for the output projection.
    const int bI = bh >> 4, h = bh & 15;
#pragma unroll
    for (int g = 0; g < 2; ++g)
#pragma unroll
        for (int r = 0; r < 8; ++r) {
            const float inv = 1.0f / li[g][r];
            const int sgl = qbase + g * 16 + r + (hsel << 3);
            const size_t rowbase = (size_t)(bI * Sc + sgl) * Dc + h * DKc + mloc;
            obuf[rowbase +  0] = f2bf(o[g][0][r] * inv);
            obuf[rowbase + 16] = f2bf(o[g][1][r] * inv);
            obuf[rowbase + 32] = f2bf(o[g][2][r] * inv);
            obuf[rowbase + 48] = f2bf(o[g][3][r] * inv);
        }
}

// ---------------------------------------------------------------------------
// Kernel 3: output projection.  out[8192,1024] = O[8192,1024] @ Wo^T (f32 out)
// 64x64 output tile per wave.
// ---------------------------------------------------------------------------
__global__ __launch_bounds__(256) void mha_oproj_wmma(
    const unsigned short* __restrict__ A, const float* __restrict__ Wo,
    float* __restrict__ out)
{
    const int lane = threadIdx.x & 31;
    const int wid  = (blockIdx.x * blockDim.x + threadIdx.x) >> 5;
    const int NT   = Dc / 64;                  // 16 column tiles
    const int row0 = (wid / NT) * 64;
    const int col0 = (wid % NT) * 64;
    const int mloc = lane & 15;
    const int hsel = (lane >> 4) & 1;

    v8f acc[16];
#pragma unroll
    for (int t = 0; t < 16; ++t) acc[t] = (v8f){0,0,0,0,0,0,0,0};

    for (int kb = 0; kb < Dc; kb += 32) {
        Frag a[4], b[4];
#pragma unroll
        for (int g = 0; g < 4; ++g) {
            const unsigned short* Ar = &A[(size_t)(row0 + g * 16 + mloc) * Dc];
            const float* Wr = &Wo[(size_t)(col0 + g * 16 + mloc) * Dc];
            __builtin_prefetch(Ar + kb + 32, 0, 0);
            __builtin_prefetch(Wr + kb + 32, 0, 0);
#pragma unroll
            for (int r = 0; r < 8; ++r) {
                const int kA = kb + ((r >> 2) << 4) + (hsel << 3) + ((r & 3) << 1);
                a[g].u[r] = *(const unsigned int*)(Ar + kA);
                const int kB = kb + ((r >> 2) << 4) + (hsel << 3) + ((r & 3) << 1);
                float2 w = *(const float2*)(Wr + kB);
                b[g].u[r] = packbf2(w.x, w.y);
            }
        }
#pragma unroll
        for (int mi = 0; mi < 4; ++mi)
#pragma unroll
            for (int ni = 0; ni < 4; ++ni)
                acc[mi * 4 + ni] = wmma_bf16(a[mi], b[ni], acc[mi * 4 + ni]);
    }

#pragma unroll
    for (int t = 0; t < 16; ++t) {
        const int rowt = row0 + ((t >> 2) << 4);
        const int colt = col0 + ((t & 3) << 4);
#pragma unroll
        for (int r = 0; r < 8; ++r) {
            const int m = rowt + r + (hsel << 3);
            const int n = colt + mloc;
            out[(size_t)m * Dc + n] = acc[t][r];
        }
    }
}

// ---------------------------------------------------------------------------
// Host-side launcher.
// Inputs: 0=x(f32), 1=mask(bool, ignored: causal computed analytically),
//         2=w_qkv(f32), 3=b_qkv(f32), 4=w_o(f32).   Output: f32 [B,S,D].
// Workspace layout (bf16 ushorts): q | k | vT | O  (8,388,608 elems each).
// ---------------------------------------------------------------------------
extern "C" void kernel_launch(void* const* d_in, const int* in_sizes, int n_in,
                              void* d_out, int out_size, void* d_ws, size_t ws_size,
                              hipStream_t stream) {
    const float* X   = (const float*)d_in[0];
    const float* Wq  = (const float*)d_in[2];
    const float* bq  = (const float*)d_in[3];
    const float* Wo  = (const float*)d_in[4];

    const size_t N = (size_t)Bc * Hc * Sc * DKc;          // 8,388,608
    unsigned short* qb  = (unsigned short*)d_ws;
    unsigned short* kb  = qb + N;
    unsigned short* vtb = kb + N;
    unsigned short* ob  = vtb + N;

    // QKV: (8192/64)*(3072/64) = 6144 wave-tiles / 8 waves per block
    mha_qkv_wmma<<<768, 256, 0, stream>>>(X, Wq, bq, qb, kb, vtb);
    // Attention: B*H*(S/32) = 4096 waves / 8 per block
    mha_attn_wmma<<<512, 256, 0, stream>>>(qb, kb, vtb, ob);
    // Output projection: (8192/64)*(1024/64) = 2048 wave-tiles / 8 per block
    mha_oproj_wmma<<<256, 256, 0, stream>>>(ob, Wo, (float*)d_out);
}
